// GraphormerLayer_71159018160674
// MI455X (gfx1250) — matrix operand order repaired and checked by reading, not compiled, not run on hardware
//
#include <hip/hip_runtime.h>
#include <hip/hip_bf16.h>
#include <math.h>

typedef __attribute__((ext_vector_type(16))) _Float16 v16h;
typedef __attribute__((ext_vector_type(8)))  _Float16 v8h;
typedef __attribute__((ext_vector_type(8)))  float    v8f;

#define D_MODEL 768
#define NHEAD   12
#define HDIM    64
#define BATCH   16
#define SEQ     512
#define MTOT    (BATCH * SEQ)   // 8192 rows
#define DFF     3072

// ---------------------------------------------------------------------------
// WMMA helpers (CDNA5 gfx1250, wave32).  D = A(16x32 f16) x B(32x16 f16) + C
// ---------------------------------------------------------------------------
__device__ __forceinline__ v8f wmma16(v16h a, v16h b, v8f c) {
  return __builtin_amdgcn_wmma_f32_16x16x32_f16(
      /*neg_a=*/false, a, /*neg_b=*/false, b,
      /*c_mod=*/(short)0, c, /*reuse_a=*/false, /*reuse_b=*/false);
}

// A-matrix 16x32 f16 fragment (ISA 7.12.2): lane l<16 -> row l, K {0..7,16..23};
// lane l>=16 -> row l-16, K {8..15,24..31}.  Source row-major, K contiguous.
// Works on global or LDS (generic) pointers.
__device__ __forceinline__ v16h load_fragA(const _Float16* __restrict__ p, size_t stride) {
  const int lane = threadIdx.x & 31;
  const _Float16* base = p + (size_t)(lane & 15) * stride + ((lane >> 4) << 3);
  v8h lo = *(const v8h*)(base);
  v8h hi = *(const v8h*)(base + 16);
  v16h f;
#pragma unroll
  for (int i = 0; i < 8; ++i) { f[i] = lo[i]; f[i + 8] = hi[i]; }
  return f;
}

// B-matrix 32x16 f16 fragment: lane n<16 -> col n, K {0..7,16..23};
// lane n>=16 -> col n-16, K {8..15,24..31}.  Source is B^T row-major (N x K, K contig).
__device__ __forceinline__ v16h load_fragB(const _Float16* __restrict__ p, size_t stride) {
  const int lane = threadIdx.x & 31;
  const _Float16* base = p + (size_t)(lane & 15) * stride + ((lane >> 4) << 3);
  v8h lo = *(const v8h*)(base);
  v8h hi = *(const v8h*)(base + 16);
  v16h f;
#pragma unroll
  for (int i = 0; i < 8; ++i) { f[i] = lo[i]; f[i + 8] = hi[i]; }
  return f;
}

// ---------------------------------------------------------------------------
// Async copy of 16 bytes global -> LDS (per lane).  VDST VGPR holds the
// wave-relative LDS byte offset (low 32 bits of the flat shared address),
// VADDR the 64-bit global address.  Tracked by ASYNCcnt.
// ---------------------------------------------------------------------------
__device__ __forceinline__ void async_ld_b128(const void* gsrc, void* lds_dst) {
  const unsigned loff = (unsigned)(size_t)lds_dst;  // flat LDS addr[31:0] = offset
  asm volatile("global_load_async_to_lds_b128 %0, %1, off"
               :
               : "v"(loff), "v"((unsigned long long)(size_t)gsrc)
               : "memory");
}
__device__ __forceinline__ void wait_async0() {
  asm volatile("s_wait_asynccnt 0x0" ::: "memory");
}

// ---------------------------------------------------------------------------
// f32 -> f16 conversion
// ---------------------------------------------------------------------------
__global__ void cvt_f32_f16_kernel(const float* __restrict__ in,
                                   _Float16* __restrict__ out, size_t n) {
  size_t i = (size_t)blockIdx.x * blockDim.x + threadIdx.x;
  const size_t stride = (size_t)gridDim.x * blockDim.x;
  for (; i < n; i += stride) out[i] = (_Float16)in[i];
}

// ---------------------------------------------------------------------------
// Generic WMMA GEMM:  C[M,N] = A[M,K] * W[N,K]^T + bias[N]  (f16 in, f32 accum)
// 256 threads = 8 waves as 4(M) x 2(N); WG tile 128x128, wave tile 32x64.
// The 128x32-half B tile is staged in LDS per K-step with async global->LDS
// copies (each B element fetched once per WG instead of once per M-wave).
// mode 0: QKV scatter (q *= 1/8 -> qh[b,h,n,d]; k -> kh[b,h,n,d]; v -> vt[b,h,d,n])
// mode 1: f32 out (ld=768)    mode 2: f16 gelu out (ld=3072)   mode 3: f32 out (ld=768)
// ---------------------------------------------------------------------------
__global__ __launch_bounds__(256)
void gemm_kernel(const _Float16* __restrict__ A, const _Float16* __restrict__ W,
                 const float* __restrict__ bias, int K, int mode,
                 float* __restrict__ outF, _Float16* __restrict__ outH,
                 _Float16* __restrict__ qh, _Float16* __restrict__ kh,
                 _Float16* __restrict__ vt) {
  __shared__ _Float16 btile[128 * 32];  // 8 KB: B tile for one K-step
  const int lane = threadIdx.x & 31;
  const int wave = threadIdx.x >> 5;
  const int m0  = blockIdx.x * 128 + (wave & 3) * 32;
  const int gn0 = blockIdx.y * 128;        // workgroup N base
  const int n0l = (wave >> 2) * 64;        // this wave's N offset inside tile

  // async-copy mapping: 512 16-byte segments = 256 threads x 2
  const int cp_n0  = threadIdx.x >> 2;          // row 0..63 (+64 on rep 1)
  const int cp_seg = (threadIdx.x & 3) << 3;    // half-offset 0,8,16,24

  const v8f vzero = {0.f, 0.f, 0.f, 0.f, 0.f, 0.f, 0.f, 0.f};
  v8f acc[2][4];
#pragma unroll
  for (int i = 0; i < 2; ++i)
#pragma unroll
    for (int j = 0; j < 4; ++j) acc[i][j] = vzero;

  for (int k0 = 0; k0 < K; k0 += 32) {
    // stage B tile (async, ASYNCcnt-tracked)
#pragma unroll
    for (int rep = 0; rep < 2; ++rep) {
      const int n = cp_n0 + rep * 64;
      async_ld_b128(W + (size_t)(gn0 + n) * K + k0 + cp_seg,
                    &btile[n * 32 + cp_seg]);
    }
    if (k0 + 32 < K)  // global_prefetch_b8 of the next A tile
      __builtin_prefetch(A + (size_t)(m0 + lane) * K + k0 + 32, 0, 3);

    v16h af[2];
#pragma unroll
    for (int i = 0; i < 2; ++i)
      af[i] = load_fragA(A + (size_t)(m0 + 16 * i) * K + k0, K);

    wait_async0();
    __syncthreads();

    v16h bf[4];
#pragma unroll
    for (int j = 0; j < 4; ++j)
      bf[j] = load_fragB(&btile[(n0l + 16 * j) * 32], 32);
#pragma unroll
    for (int i = 0; i < 2; ++i)
#pragma unroll
      for (int j = 0; j < 4; ++j)
        acc[i][j] = wmma16(af[i], bf[j], acc[i][j]);
    __syncthreads();  // tile consumed before next-step overwrite
  }

  // Epilogue.  C layout: VGPR r, lanes 0-15 -> (row r, col lane); lanes 16-31 -> (row r+8).
  const int rowadd = (lane >> 4) << 3;
#pragma unroll
  for (int i = 0; i < 2; ++i) {
#pragma unroll
    for (int j = 0; j < 4; ++j) {
      const int col = gn0 + n0l + 16 * j + (lane & 15);
      const float bc = bias[col];
#pragma unroll
      for (int r = 0; r < 8; ++r) {
        const int row = m0 + 16 * i + r + rowadd;
        const float v = acc[i][j][r] + bc;
        if (mode == 0) {
          const int s = col / D_MODEL;
          const int rem = col - s * D_MODEL;
          const int h = rem >> 6, d = rem & 63;
          const int b = row >> 9, nq = row & 511;
          const size_t bh = (size_t)(b * NHEAD + h);
          if (s == 0)      qh[(bh * SEQ + nq) * HDIM + d] = (_Float16)(v * 0.125f);
          else if (s == 1) kh[(bh * SEQ + nq) * HDIM + d] = (_Float16)v;
          else             vt[(bh * HDIM + d) * SEQ + nq] = (_Float16)v;
        } else if (mode == 2) {
          const float g = 0.5f * v * (1.0f + erff(v * 0.70710678118654752f));
          outH[(size_t)row * DFF + col] = (_Float16)g;
        } else {
          outF[(size_t)row * D_MODEL + col] = v;
        }
      }
    }
  }
}

// ---------------------------------------------------------------------------
// Attention: one wave (32 threads) per (b, h, 16-query tile).
// Phase 1: scores(16x512) = q*k^T/8 + bias (2 WMMAs / 16 keys) -> LDS f32.
// Phase 2: row max / exp / sum (2 lanes per row, shfl-combined); exp written
//          as f16 into a second LDS region laid out for A-fragment ds_load_b128.
// Phase 3: out(16x64) = exp_scores @ V^T via WMMA, 1/sum in the epilogue,
//          stored f16 head-interleaved as [b*512+q, 768] for the out-proj GEMM.
// Dynamic LDS: 16*512*4 + 16*512*2 = 48 KB.
// ---------------------------------------------------------------------------
__global__ __launch_bounds__(32)
void attn_kernel(const _Float16* __restrict__ qh, const _Float16* __restrict__ kh,
                 const _Float16* __restrict__ vt, const float* __restrict__ bias,
                 const unsigned char* __restrict__ mask, _Float16* __restrict__ ao) {
  extern __shared__ char smem_raw[];
  float*    sc = (float*)smem_raw;                          // 16 x 512 f32
  _Float16* ax = (_Float16*)(smem_raw + 16 * SEQ * 4);      // 16 x 512 f16 (exp)

  const int lane = threadIdx.x & 31;
  const int qblk = blockIdx.x & 31;
  const int bh = blockIdx.x >> 5;
  const int b = bh / NHEAD;
  const int h = bh - b * NHEAD;
  const int q0 = qblk * 16;

  const _Float16* qbase = qh + ((size_t)bh * SEQ + q0) * HDIM;
  const _Float16* kbase = kh + (size_t)bh * SEQ * HDIM;
  const float* bbase = bias + ((size_t)bh * SEQ + q0) * SEQ;

  const v16h qa0 = load_fragA(qbase, HDIM);        // K = d 0..31 (q pre-scaled)
  const v16h qa1 = load_fragA(qbase + 32, HDIM);   // K = d 32..63
  const int rowadd = (lane >> 4) << 3;
  const v8f vzero = {0.f, 0.f, 0.f, 0.f, 0.f, 0.f, 0.f, 0.f};

  for (int n0 = 0; n0 < SEQ; n0 += 16) {
    const v16h kb0 = load_fragB(kbase + (size_t)n0 * HDIM, HDIM);
    const v16h kb1 = load_fragB(kbase + (size_t)n0 * HDIM + 32, HDIM);
    v8f c = vzero;
    c = wmma16(qa0, kb0, c);
    c = wmma16(qa1, kb1, c);
    const int col = n0 + (lane & 15);
    const bool ok = mask[b * SEQ + col] != 0;
#pragma unroll
    for (int r = 0; r < 8; ++r) {
      const int rl = r + rowadd;
      const float v = c[r] + bbase[(size_t)rl * SEQ + col];
      sc[rl * SEQ + col] = ok ? v : -3.0e38f;
    }
  }
  __syncthreads();

  // Softmax statistics: lane pair (l, l+16) handles row l&15, 256 cols each.
  const int rl = lane & 15;
  const int part = lane >> 4;
  const float* rowp = sc + rl * SEQ + part * 256;
  _Float16*    rowx = ax + rl * SEQ + part * 256;
  float mx = -3.0e38f;
  for (int c = 0; c < 256; ++c) mx = fmaxf(mx, rowp[c]);
  mx = fmaxf(mx, __shfl_xor(mx, 16, 32));
  float sum = 0.f;
  for (int c = 0; c < 256; ++c) {
    const float e = __expf(rowp[c] - mx);
    rowx[c] = (_Float16)e;  // disjoint per-lane ranges: no aliasing
    sum += e;
  }
  sum += __shfl_xor(sum, 16, 32);
  const float rs = (sum > 0.f) ? (1.0f / sum) : 0.f;
  __syncthreads();

  // out = exp_scores(16x512) @ V^T(512x64); A-fragments via ds_load_b128 pairs.
  const _Float16* vbase = vt + (size_t)bh * HDIM * SEQ;
  v8f acc[4] = {vzero, vzero, vzero, vzero};
  for (int ks = 0; ks < SEQ; ks += 32) {
    const v16h a = load_fragA(ax + ks, SEQ);
#pragma unroll
    for (int j = 0; j < 4; ++j) {
      const v16h bv = load_fragB(vbase + (size_t)(j * 16) * SEQ + ks, SEQ);
      acc[j] = wmma16(a, bv, acc[j]);
    }
  }
#pragma unroll
  for (int j = 0; j < 4; ++j) {
    const int d = j * 16 + (lane & 15);
#pragma unroll
    for (int r = 0; r < 8; ++r) {
      const int rq = r + rowadd;
      const float s = __shfl(rs, rq, 32);
      ao[(size_t)(b * SEQ + q0 + rq) * D_MODEL + h * HDIM + d] =
          (_Float16)(acc[j][r] * s);
    }
  }
}

// ---------------------------------------------------------------------------
// out = LayerNorm(xa + xb) * gamma + beta ; writes f32 (residual) and/or f16
// ---------------------------------------------------------------------------
__global__ __launch_bounds__(256)
void add_ln_kernel(const float* __restrict__ xa, const float* __restrict__ xb,
                   const float* __restrict__ gamma, const float* __restrict__ beta,
                   float* __restrict__ outF, _Float16* __restrict__ outH) {
  __shared__ float ssum[256];
  __shared__ float ssq[256];
  const int row = blockIdx.x;
  const int t = threadIdx.x;
  const float* pa = xa + (size_t)row * D_MODEL;
  const float* pb = xb + (size_t)row * D_MODEL;
  float v[3];
  float s = 0.f, q = 0.f;
#pragma unroll
  for (int i = 0; i < 3; ++i) {
    const int c = t + i * 256;
    v[i] = pa[c] + pb[c];
    s += v[i];
    q += v[i] * v[i];
  }
  ssum[t] = s; ssq[t] = q;
  __syncthreads();
  for (int o = 128; o > 0; o >>= 1) {
    if (t < o) { ssum[t] += ssum[t + o]; ssq[t] += ssq[t + o]; }
    __syncthreads();
  }
  const float mu = ssum[0] * (1.0f / D_MODEL);
  const float var = ssq[0] * (1.0f / D_MODEL) - mu * mu;
  const float rstd = rsqrtf(var + 1e-5f);
#pragma unroll
  for (int i = 0; i < 3; ++i) {
    const int c = t + i * 256;
    const float o = (v[i] - mu) * rstd * gamma[c] + beta[c];
    if (outF) outF[(size_t)row * D_MODEL + c] = o;
    if (outH) outH[(size_t)row * D_MODEL + c] = (_Float16)o;
  }
}

// ---------------------------------------------------------------------------
// Launch
// ---------------------------------------------------------------------------
extern "C" void kernel_launch(void* const* d_in, const int* in_sizes, int n_in,
                              void* d_out, int out_size, void* d_ws, size_t ws_size,
                              hipStream_t stream) {
  (void)in_sizes; (void)n_in; (void)out_size; (void)ws_size;
  const float* x     = (const float*)d_in[0];
  const float* abias = (const float*)d_in[1];
  const unsigned char* mask = (const unsigned char*)d_in[2];
  const float* qkv_w = (const float*)d_in[3];
  const float* qkv_b = (const float*)d_in[4];
  const float* out_w = (const float*)d_in[5];
  const float* out_b = (const float*)d_in[6];
  const float* ln1_s = (const float*)d_in[7];
  const float* ln1_b = (const float*)d_in[8];
  const float* ln2_s = (const float*)d_in[9];
  const float* ln2_b = (const float*)d_in[10];
  const float* ff1_w = (const float*)d_in[11];
  const float* ff1_b = (const float*)d_in[12];
  const float* ff2_w = (const float*)d_in[13];
  const float* ff2_b = (const float*)d_in[14];
  float* out = (float*)d_out;

  char* ws = (char*)d_ws;
  size_t off = 0;
  auto alloc = [&](size_t bytes) -> char* {
    char* p = ws + off;
    off = (off + bytes + 255) & ~(size_t)255;
    return p;
  };
  const size_t NE = (size_t)MTOT * D_MODEL;  // 6291456
  _Float16* xh    = (_Float16*)alloc(NE * 2);
  _Float16* qkvwh = (_Float16*)alloc((size_t)3 * D_MODEL * D_MODEL * 2);
  _Float16* outwh = (_Float16*)alloc((size_t)D_MODEL * D_MODEL * 2);
  _Float16* ff1wh = (_Float16*)alloc((size_t)DFF * D_MODEL * 2);
  _Float16* ff2wh = (_Float16*)alloc((size_t)D_MODEL * DFF * 2);
  _Float16* qH    = (_Float16*)alloc(NE * 2);
  _Float16* kH    = (_Float16*)alloc(NE * 2);
  _Float16* vT    = (_Float16*)alloc(NE * 2);
  _Float16* aoH   = (_Float16*)alloc(NE * 2);
  float*    yF    = (float*)alloc(NE * 4);
  float*    x1F   = (float*)alloc(NE * 4);
  _Float16* x1H   = (_Float16*)alloc(NE * 2);
  _Float16* hH    = (_Float16*)alloc((size_t)MTOT * DFF * 2);
  float*    f2F   = (float*)alloc(NE * 4);

  auto cvt = [&](const float* src, _Float16* dst, size_t n) {
    int blocks = (int)((n + 2047) / 2048);
    if (blocks > 4096) blocks = 4096;
    cvt_f32_f16_kernel<<<blocks, 256, 0, stream>>>(src, dst, n);
  };
  cvt(x, xh, NE);
  cvt(qkv_w, qkvwh, (size_t)3 * D_MODEL * D_MODEL);
  cvt(out_w, outwh, (size_t)D_MODEL * D_MODEL);
  cvt(ff1_w, ff1wh, (size_t)DFF * D_MODEL);
  cvt(ff2_w, ff2wh, (size_t)D_MODEL * DFF);

  // QKV projection (M=8192, K=768, N=2304)
  gemm_kernel<<<dim3(MTOT / 128, 2304 / 128), 256, 0, stream>>>(
      xh, qkvwh, qkv_b, D_MODEL, 0, nullptr, nullptr, qH, kH, vT);

  // Attention: 16*12*32 = 6144 single-wave blocks, 48 KB dynamic LDS each.
  attn_kernel<<<BATCH * NHEAD * (SEQ / 16), 32,
                16 * SEQ * sizeof(float) + 16 * SEQ * sizeof(_Float16), stream>>>(
      qH, kH, vT, abias, mask, aoH);

  // Output projection (N=768) -> f32
  gemm_kernel<<<dim3(MTOT / 128, D_MODEL / 128), 256, 0, stream>>>(
      aoH, outwh, out_b, D_MODEL, 1, yF, nullptr, nullptr, nullptr, nullptr);

  // x1 = LN(x + attn_out)
  add_ln_kernel<<<MTOT, 256, 0, stream>>>(x, yF, ln1_s, ln1_b, x1F, x1H);

  // FF1 + GELU (N=3072) -> f16
  gemm_kernel<<<dim3(MTOT / 128, DFF / 128), 256, 0, stream>>>(
      x1H, ff1wh, ff1_b, D_MODEL, 2, nullptr, hH, nullptr, nullptr, nullptr);

  // FF2 (K=3072, N=768) -> f32
  gemm_kernel<<<dim3(MTOT / 128, D_MODEL / 128), 256, 0, stream>>>(
      hH, ff2wh, ff2_b, DFF, 3, f2F, nullptr, nullptr, nullptr, nullptr);

  // out = LN(x1 + ff_out)
  add_ln_kernel<<<MTOT, 256, 0, stream>>>(x1F, f2F, ln2_s, ln2_b, out, nullptr);
}
